// CrfLayer_63522566308155
// MI455X (gfx1250) — compile-verified, matrix operand order checked by reading, not run on hardware
//
#include <hip/hip_runtime.h>

#define B_ 256
#define T_ 512
#define N_ 128

typedef __attribute__((ext_vector_type(16))) _Float16     v16h;
typedef __attribute__((ext_vector_type(8)))  float        v8f;
typedef __attribute__((ext_vector_type(8)))  unsigned int v8u;

union HU { v8u u; v16h h; };

// K-index packing for 16-bit WMMA operands (ISA 7.12.2, 16-bit A 16x32):
// VGPR v holds K = 16*(v>>2) + 8*h + 2*(v&3) + {0,1}, h = lane>>4.
__device__ __forceinline__ int kpack(int v, int h) {
  return 16 * (v >> 2) + 8 * h + 2 * (v & 3);
}

// Async global -> LDS copy, 16 bytes per lane (GLOBAL_LOAD_ASYNC_TO_LDS_B128,
// GV addressing; tracked by ASYNCcnt). lds_off = byte offset within the
// workgroup LDS allocation (low 32 bits of the generic shared pointer).
__device__ __forceinline__ void async_ld_b128(unsigned lds_off, const float* g) {
  asm volatile("global_load_async_to_lds_b128 %0, %1, off"
               :: "v"(lds_off), "v"(g)
               : "memory");
}

// ---------------------------------------------------------------------------
// trans passthrough output
// ---------------------------------------------------------------------------
extern "C" __global__ __launch_bounds__(256)
void crf_copy_trans(const float* __restrict__ trans, float* __restrict__ out_trans) {
  int i = blockIdx.x * 256 + threadIdx.x;
  if (i < N_ * N_) out_trans[i] = trans[i];
}

// ---------------------------------------------------------------------------
// Forward pass: log-norm recursion via WMMA (real-semiring matvec after
// max-shift) fused with Viterbi max/argmax recursion on VALU.
// One block handles 16 batch rows; 8 waves, wave w owns N-tile [16w,16w+16).
// Emissions for step t+1 are prefetched into LDS with async b128 loads.
// ---------------------------------------------------------------------------
extern "C" __global__ __launch_bounds__(256)
void crf_forward_kernel(const float* __restrict__ emissions,
                        const int* __restrict__ mask,
                        const float* __restrict__ trans,
                        unsigned char* __restrict__ bp,
                        float* __restrict__ lognorm,
                        int* __restrict__ lasttag)
{
  extern __shared__ char smem[];
  float*    trans_s  = (float*)(smem);             // 128*128*4 = 65536
  float*    alpha_s  = (float*)(smem + 65536);     // 16*128*4  =  8192
  float*    valpha_s = (float*)(smem + 73728);     // 16*128*4  =  8192
  float*    emitb[2];                               // double-buffered emissions
  emitb[0]           = (float*)(smem + 81920);     // 8192
  emitb[1]           = (float*)(smem + 90112);     // 8192
  _Float16* ap_s     = (_Float16*)(smem + 98304);  // 16*128*2  =  4096
  float*    rowred   = (float*)(smem + 102400);    // 16*16*4   =  1024
  float*    m_s      = (float*)(smem + 103424);    // 16*4
  float*    mrow     = (float*)(smem + 103488);    // 16*4

  const int tid    = threadIdx.x;
  const int rb     = blockIdx.x * 16;    // first batch row of this block
  const int lane   = tid & 31;
  const int w      = tid >> 5;           // wave id = N-tile id
  const int lr     = lane & 15;
  const int lh     = lane >> 4;
  const int nw     = w * 16 + lr;        // this lane's output column (0..127)
  const int lane_j = tid & 127;          // viterbi column
  const int rbase  = tid >> 7;           // 0/1: rows rbase+2q, q=0..7

  // Load trans (f32) into LDS once.
  for (int i = tid; i < N_ * N_; i += 256) trans_s[i] = trans[i];

  // alpha0 = valpha0 = emissions[:,0,:]
  for (int q = 0; q < 8; ++q) {
    int r = rbase + 2 * q;
    float e = emissions[((size_t)(rb + r) * T_ + 0) * N_ + lane_j];
    alpha_s[r * N_ + lane_j]  = e;
    valpha_s[r * N_ + lane_j] = e;
  }

  // Kick off async prefetch of emissions for t=1 into buffer 1.
  {
    int f = tid, r = f >> 5, c4 = (f & 31) * 4;
    async_ld_b128((unsigned)(size_t)(emitb[1] + r * N_ + c4),
                  &emissions[((size_t)(rb + r) * T_ + 1) * N_ + c4]);
    f = tid + 256; r = f >> 5; c4 = (f & 31) * 4;
    async_ld_b128((unsigned)(size_t)(emitb[1] + r * N_ + c4),
                  &emissions[((size_t)(rb + r) * T_ + 1) * N_ + c4]);
  }
  __syncthreads();

  // Preload B fragments: E = exp(trans), column nw. B mirrors A's packing
  // with N = lane&15 (ISA 7.12.2). Held in registers for the whole T loop.
  v16h Bfrag[4];
  for (int kc = 0; kc < 4; ++kc) {
    v16h bf;
#pragma unroll
    for (int v = 0; v < 8; ++v) {
      int k0 = kc * 32 + kpack(v, lh);
      bf[2 * v]     = (_Float16)__expf(trans_s[(k0)     * N_ + nw]);
      bf[2 * v + 1] = (_Float16)__expf(trans_s[(k0 + 1) * N_ + nw]);
    }
    Bfrag[kc] = bf;
  }

  const unsigned int* apu = (const unsigned int*)ap_s;

  for (int t = 1; t < T_; ++t) {
    __syncthreads();  // prior alpha/valpha writes + async-buffer reads retired
    const float* emit_s = emitb[t & 1];

    // ---- Phase A: mask, async prefetch t+1, row-max partials -------------
    if (tid < 16) mrow[tid] = (float)mask[(rb + tid) * T_ + t];
    if (t + 1 < T_) {
      float* nb = emitb[(t + 1) & 1];
      int f = tid, r = f >> 5, c4 = (f & 31) * 4;
      async_ld_b128((unsigned)(size_t)(nb + r * N_ + c4),
                    &emissions[((size_t)(rb + r) * T_ + (t + 1)) * N_ + c4]);
      f = tid + 256; r = f >> 5; c4 = (f & 31) * 4;
      async_ld_b128((unsigned)(size_t)(nb + r * N_ + c4),
                    &emissions[((size_t)(rb + r) * T_ + (t + 1)) * N_ + c4]);
    }
    {
      int r = tid >> 4, c = tid & 15;
      float mx = -3.0e38f;
#pragma unroll
      for (int k = 0; k < 8; ++k) {
        float v = alpha_s[r * N_ + c * 8 + k];
        mx = v > mx ? v : mx;
      }
      rowred[r * 16 + c] = mx;
    }
    __syncthreads();
    if (tid < 16) {
      float mx = rowred[tid * 16];
#pragma unroll
      for (int c = 1; c < 16; ++c) {
        float v = rowred[tid * 16 + c];
        mx = v > mx ? v : mx;
      }
      m_s[tid] = mx;
    }
    __syncthreads();

    // ---- Phase B: p = exp(alpha - m) as f16 A-matrix ---------------------
    for (int q = 0; q < 8; ++q) {
      int r = rbase + 2 * q;
      ap_s[r * N_ + lane_j] =
          (_Float16)__expf(alpha_s[r * N_ + lane_j] - m_s[r]);
    }
    __syncthreads();

    // ---- Phase C1: WMMA s = p x exp(trans), K=128 in 4 chunks ------------
    v8f cf = {};
#pragma unroll
    for (int kc = 0; kc < 4; ++kc) {
      HU a;
#pragma unroll
      for (int v = 0; v < 8; ++v) {
        int k0 = kc * 32 + kpack(v, lh);          // even -> aligned b32 pair
        a.u[v] = apu[(lr * N_ + k0) >> 1];
      }
      cf = __builtin_amdgcn_wmma_f32_16x16x32_f16(
          false, a.h, false, Bfrag[kc], (short)0, cf, false, false);
    }

    // ---- Phase C2: Viterbi max/argmax (tropical, VALU) -------------------
    float vmax[8];
    int   varg[8];
#pragma unroll
    for (int q = 0; q < 8; ++q) { vmax[q] = -3.0e38f; varg[q] = 0; }
    for (int i = 0; i < N_; ++i) {
      float tr = trans_s[i * N_ + lane_j];        // conflict-free across lanes
#pragma unroll
      for (int q = 0; q < 8; ++q) {
        float v = valpha_s[(rbase + 2 * q) * N_ + i] + tr;  // LDS broadcast
        if (v > vmax[q]) { vmax[q] = v; varg[q] = i; }
      }
    }

    // Ensure this step's emissions (issued last iteration) have landed:
    // 2 async issues/wave/step, in-order completion => <=2 outstanding means
    // the older pair is done. Final step has nothing newer in flight.
    if (t + 1 < T_) asm volatile("s_wait_asynccnt 0x2" ::: "memory");
    else            asm volatile("s_wait_asynccnt 0x0" ::: "memory");
    __syncthreads();

    // ---- Phase D: commit new alpha (C frag), valpha, backpointers --------
#pragma unroll
    for (int i = 0; i < 8; ++i) {
      int row = i + 8 * lh;                        // C/D layout (ISA 7.12.2)
      if (mrow[row] > 0.f)
        alpha_s[row * N_ + nw] =
            m_s[row] + __logf(cf[i]) + emit_s[row * N_ + nw];
    }
#pragma unroll
    for (int q = 0; q < 8; ++q) {
      int r = rbase + 2 * q;
      unsigned char bv;
      if (mrow[r] > 0.f) {
        valpha_s[r * N_ + lane_j] = vmax[q] + emit_s[r * N_ + lane_j];
        bv = (unsigned char)varg[q];
      } else {
        bv = (unsigned char)lane_j;               // bp = idx when masked
      }
      bp[((size_t)(rb + r) * T_ + t) * N_ + lane_j] = bv;
    }
  }
  __syncthreads();

  // ---- Finalize: log-norm = logsumexp(alpha), last_tag = argmax(valpha) --
  if (tid < 16) {
    int r = tid;
    float mx = -3.0e38f;
    for (int j = 0; j < N_; ++j) { float v = alpha_s[r * N_ + j]; mx = v > mx ? v : mx; }
    float s = 0.f;
    for (int j = 0; j < N_; ++j) s += __expf(alpha_s[r * N_ + j] - mx);
    lognorm[rb + r] = mx + __logf(s);
    float bvv = valpha_s[r * N_];
    int bi = 0;
    for (int j = 1; j < N_; ++j) {
      float v = valpha_s[r * N_ + j];
      if (v > bvv) { bvv = v; bi = j; }
    }
    lasttag[rb + r] = bi;
  }
}

// ---------------------------------------------------------------------------
// Gold-path sequence score + log-likelihood
// ---------------------------------------------------------------------------
extern "C" __global__ __launch_bounds__(256)
void crf_score_kernel(const float* __restrict__ emissions,
                      const int* __restrict__ tag,
                      const int* __restrict__ mask,
                      const float* __restrict__ trans,
                      const float* __restrict__ lognorm,
                      float* __restrict__ out_ll)
{
  int b = blockIdx.x * blockDim.x + threadIdx.x;
  if (b >= B_) return;
  float sc = 0.f;
  int prev = 0;
  for (int t = 0; t < T_; ++t) {
    int tg = tag[b * T_ + t];
    float mk = (float)mask[b * T_ + t];
    sc += emissions[((size_t)b * T_ + t) * N_ + tg] * mk;
    if (t > 0) sc += trans[prev * N_ + tg] * mk;
    prev = tg;
  }
  out_ll[b] = sc - lognorm[b];
}

// ---------------------------------------------------------------------------
// Viterbi backtrace: one thread per sequence
// ---------------------------------------------------------------------------
extern "C" __global__ __launch_bounds__(256)
void crf_backward_kernel(const unsigned char* __restrict__ bp,
                         const int* __restrict__ lasttag,
                         float* __restrict__ out_pred)
{
  int b = blockIdx.x * blockDim.x + threadIdx.x;
  if (b >= B_) return;
  int tg = lasttag[b];
  out_pred[(size_t)b * T_ + (T_ - 1)] = (float)tg;
  for (int t = T_ - 1; t >= 1; --t) {
    tg = bp[((size_t)b * T_ + t) * N_ + tg];
    out_pred[(size_t)b * T_ + (t - 1)] = (float)tg;
  }
}

// ---------------------------------------------------------------------------
extern "C" void kernel_launch(void* const* d_in, const int* in_sizes, int n_in,
                              void* d_out, int out_size, void* d_ws, size_t ws_size,
                              hipStream_t stream)
{
  const float* emissions = (const float*)d_in[0];
  const int*   tag       = (const int*)d_in[1];
  const int*   mask      = (const int*)d_in[2];
  const float* trans     = (const float*)d_in[3];

  float* out       = (float*)d_out;
  float* out_ll    = out;                       // [256]
  float* out_trans = out + B_;                  // [16384]
  float* out_pred  = out + B_ + N_ * N_;        // [256*512]

  // Workspace: bp u8 [B,T,N] (16 MiB), lognorm f32 [B], lasttag i32 [B]
  unsigned char* bp = (unsigned char*)d_ws;
  float* lognorm = (float*)((char*)d_ws + (size_t)B_ * T_ * N_);
  int*   lasttag = (int*)((char*)d_ws + (size_t)B_ * T_ * N_ + B_ * sizeof(float));

  const size_t SMEM = 103552;  // trans(64K)+alpha/valpha(16K)+emit x2(16K)+p16+red
  (void)hipFuncSetAttribute((const void*)crf_forward_kernel,
                            hipFuncAttributeMaxDynamicSharedMemorySize,
                            (int)SMEM);

  crf_copy_trans<<<dim3((N_ * N_ + 255) / 256), dim3(256), 0, stream>>>(trans, out_trans);
  crf_forward_kernel<<<dim3(B_ / 16), dim3(256), SMEM, stream>>>(
      emissions, mask, trans, bp, lognorm, lasttag);
  crf_score_kernel<<<dim3(1), dim3(256), 0, stream>>>(
      emissions, tag, mask, trans, lognorm, out_ll);
  crf_backward_kernel<<<dim3(1), dim3(256), 0, stream>>>(bp, lasttag, out_pred);
}
